// DIN_47545287967429
// MI455X (gfx1250) — matrix operand order repaired and checked by reading, not compiled, not run on hardware
//
#include <hip/hip_runtime.h>
#include <hip/hip_bf16.h>

// ---------------------------------------------------------------------------
// DIN forward (B=16384, L=50, D=32) for gfx1250:
//   Kernel 1: embedding gather + attention (WMMA f16) + softmax + weighted sum
//             -> combined[B,160] f16 scratch in d_ws
//   Kernel 2: 160->256->128->1 MLP (WMMA f16 for layers 1-2)
// ---------------------------------------------------------------------------

typedef _Float16 h2   __attribute__((ext_vector_type(2)));
typedef _Float16 v16h __attribute__((ext_vector_type(16)));
typedef float    v8f  __attribute__((ext_vector_type(8)));

union AFrag { v16h v; h2 p[8]; };

#define LL 50   // history length
#define DD 32   // embedding dim

// ---- LDS layout constants, kernel 1 ---------------------------------------
// aw1t  [80][256] f16 : 40960 B   (shared per WG)
// ab1f  [80]  f32     :   320 B
// aw2f  [80]  f32     :   320 B
// per wave (8 waves):
//   hist [64][64] f16 : 8192 B
//   qv   [64]     f16 :  128 B
//   sw   [64]     f32 :  256 B    -> 8576 B/wave
#define K1_WAVE_BASE 41600
#define K1_WAVE_SZ   8576
#define K1_LDS_BYTES (K1_WAVE_BASE + 8 * K1_WAVE_SZ)   // 110208

__global__ __launch_bounds__(256)
void din_attention_kernel(const int* __restrict__ customer_id,
                          const int* __restrict__ cand_good,
                          const int* __restrict__ cand_class,
                          const int* __restrict__ hist_goods,
                          const int* __restrict__ hist_classes,
                          const float* __restrict__ user_table,
                          const float* __restrict__ item_table,
                          const float* __restrict__ cat_table,
                          const float* __restrict__ aw1,   // [256][80]
                          const float* __restrict__ ab1,   // [80]
                          const float* __restrict__ aw2,   // [80]
                          _Float16* __restrict__ combined, // [B][160]
                          int B)
{
    extern __shared__ char smem[];
    _Float16* aw1t = (_Float16*)smem;                 // [80][256] transposed (K contiguous)
    float*    ab1f = (float*)(smem + 40960);
    float*    aw2f = (float*)(smem + 41280);

    const int tid = threadIdx.x;

    // ---- cooperative staging of attention weights (transposed, f16) ----
    for (int i = tid; i < 80 * 256; i += 256) {
        int n = i >> 8, k = i & 255;                  // aw1t[n][k] = aw1[k][n]
        aw1t[i] = (_Float16)aw1[k * 80 + n];
    }
    for (int i = tid; i < 80; i += 256) { ab1f[i] = ab1[i]; aw2f[i] = aw2[i]; }
    __syncthreads();

    const int wid  = tid >> 5;
    const int lane = tid & 31;
    const bool hiL = lane >= 16;
    const int ln   = lane & 15;

    const int b = blockIdx.x * 8 + wid;
    if (b >= B) return;

    char* wb = smem + K1_WAVE_BASE + wid * K1_WAVE_SZ;
    _Float16* hist = (_Float16*)wb;                   // [64][64]
    _Float16* qv   = (_Float16*)(wb + 8192);          // [64]
    float*    sw   = (float*)(wb + 8320);             // [64]

    // ---- gather embeddings ----
    const int cg = cand_good[b];
    const int cc = cand_class[b];
    const float qi = item_table[(size_t)cg * DD + lane];
    const float qc = cat_table[(size_t)cc * DD + lane];
    qv[lane]      = (_Float16)qi;
    qv[32 + lane] = (_Float16)qc;

    _Float16* cb = combined + (size_t)b * 160;
    cb[lane]      = (_Float16)user_table[(size_t)customer_id[b] * DD + lane];
    cb[32 + lane] = (_Float16)qi;
    cb[64 + lane] = (_Float16)qc;

    const int* hg = hist_goods   + (size_t)b * LL;
    const int* hc = hist_classes + (size_t)b * LL;
    for (int l = 0; l < LL; ++l) {
        if (l + 8 < LL) {   // warm L2/L0 for upcoming gather rows
            __builtin_prefetch(item_table + (size_t)hg[l + 8] * DD, 0, 0);
            __builtin_prefetch(cat_table  + (size_t)hc[l + 8] * DD, 0, 0);
        }
        hist[l * 64 + lane]      = (_Float16)item_table[(size_t)hg[l] * DD + lane];
        hist[l * 64 + 32 + lane] = (_Float16)cat_table [(size_t)hc[l] * DD + lane];
    }
    for (int l = LL; l < 64; ++l) {
        hist[l * 64 + lane] = (_Float16)0.f;
        hist[l * 64 + 32 + lane] = (_Float16)0.f;
    }

    // ---- attention: h = relu(info @ aw1 + ab1); scores = h @ aw2 ----
    // info columns: [0,64)=q  [64,128)=hist  [128,192)=q-hist  [192,256)=q*hist
    const h2* qp = (const h2*)qv;
    for (int mt = 0; mt < 4; ++mt) {
        const int r = mt * 16 + ln;                   // A-matrix row for this lane
        const h2* hr = (const h2*)(hist + r * 64);
        float sc[8] = {0.f, 0.f, 0.f, 0.f, 0.f, 0.f, 0.f, 0.f};

        for (int nt = 0; nt < 5; ++nt) {
            v8f acc = {};
            for (int kk = 0; kk < 8; ++kk) {
                const int seg = kk >> 1;              // which info segment
                const int cw0 = (kk & 1) * 32;        // 0 or 32 within segment
                AFrag a, bf;
#pragma unroll
                for (int v = 0; v < 8; ++v) {
                    // 16-bit A layout: dword v -> K pair starting at:
                    const int k0 = 2 * v + (v >= 4 ? 8 : 0) + (hiL ? 8 : 0);
                    const int p  = (cw0 + k0) >> 1;   // h2 index within segment
                    h2 val;
                    if (seg == 0)      val = qp[p];
                    else if (seg == 1) val = hr[p];
                    else if (seg == 2) val = qp[p] - hr[p];
                    else               val = qp[p] * hr[p];
                    a.p[v] = val;
                }
                const int n = nt * 16 + ln;
                const h2* wp = (const h2*)(aw1t + n * 256 + kk * 32);
#pragma unroll
                for (int v = 0; v < 8; ++v) {
                    // 16-bit B layout: dword v -> K pair 2v (+8 for v>=4, +8 for hi lanes)
                    bf.p[v] = wp[v + (v >= 4 ? 4 : 0) + (hiL ? 4 : 0)];
                }
                acc = __builtin_amdgcn_wmma_f32_16x16x32_f16(
                          false, a.v, false, bf.v, (short)0, acc, false, false);
            }
            const float bias = ab1f[nt * 16 + ln];
            const float w2   = aw2f[nt * 16 + ln];
#pragma unroll
            for (int v = 0; v < 8; ++v) {
                float hv = acc[v] + bias;
                hv = hv > 0.f ? hv : 0.f;             // relu
                sc[v] += hv * w2;                     // fold in h @ aw2
            }
        }
        // reduce the 16 N-lanes of each group; D row = v + (hi?8:0)
#pragma unroll
        for (int v = 0; v < 8; ++v) {
            float s = sc[v];
            s += __shfl_xor(s, 8, 32);
            s += __shfl_xor(s, 4, 32);
            s += __shfl_xor(s, 2, 32);
            s += __shfl_xor(s, 1, 32);
            if (ln == 0) sw[mt * 16 + (hiL ? 8 : 0) + v] = s;
        }
    }

    // ---- masked softmax over L (each lane owns rows lane, lane+32) ----
    float s0 = sw[lane];
    float s1 = sw[lane + 32];
    if (lane >= LL || hg[lane] == 0) s0 = -1e9f;
    if (lane + 32 >= LL || (lane + 32 < LL && hg[lane + 32] == 0)) s1 = -1e9f;
    float m = fmaxf(s0, s1);
    for (int off = 16; off >= 1; off >>= 1) m = fmaxf(m, __shfl_xor(m, off, 32));
    const float e0 = __expf(s0 - m);
    const float e1 = __expf(s1 - m);
    float sum = e0 + e1;
    for (int off = 16; off >= 1; off >>= 1) sum += __shfl_xor(sum, off, 32);
    const float inv = 1.f / sum;
    sw[lane]      = e0 * inv;
    sw[lane + 32] = e1 * inv;

    // ---- att = weights @ hist ----
    float a0 = 0.f, a1 = 0.f;
    for (int l = 0; l < LL; ++l) {
        const float w = sw[l];
        a0 += w * (float)hist[l * 64 + lane];
        a1 += w * (float)hist[l * 64 + 32 + lane];
    }
    cb[96 + lane]  = (_Float16)a0;
    cb[128 + lane] = (_Float16)a1;
}

// ---- LDS layout constants, kernel 2 ---------------------------------------
// A0 [64][160] f16 : 20480
// W  (weights, transposed, f16) : 81920  (max = mw1t [256][160])
// Z1 [64][256] f16 : 32768
// Z2 [64][128] f16 : 16384
// w3 [128] f32     :   512
#define K2_OFF_W   20480
#define K2_OFF_Z1  (20480 + 81920)
#define K2_OFF_Z2  (20480 + 81920 + 32768)
#define K2_OFF_W3  (20480 + 81920 + 32768 + 16384)
#define K2_LDS_BYTES (K2_OFF_W3 + 512)                // 152064

__global__ __launch_bounds__(128)
void din_mlp_kernel(const _Float16* __restrict__ combined, // [B][160]
                    const float* __restrict__ mw1,          // [160][256]
                    const float* __restrict__ mb1,          // [256]
                    const float* __restrict__ mw2,          // [256][128]
                    const float* __restrict__ mb2,          // [128]
                    const float* __restrict__ mw3,          // [128]
                    const float* __restrict__ mb3,          // [1]
                    float* __restrict__ out, int B)
{
    extern __shared__ char smem[];
    _Float16* A0 = (_Float16*)smem;
    _Float16* W  = (_Float16*)(smem + K2_OFF_W);
    _Float16* Z1 = (_Float16*)(smem + K2_OFF_Z1);
    _Float16* Z2 = (_Float16*)(smem + K2_OFF_Z2);
    float*    w3 = (float*)(smem + K2_OFF_W3);

    const int tid  = threadIdx.x;
    const int row0 = blockIdx.x * 64;

    // stage activations + mw1 transposed [256][160]
    for (int i = tid; i < 64 * 160; i += 128)
        A0[i] = combined[(size_t)row0 * 160 + i];
    for (int i = tid; i < 256 * 160; i += 128) {
        int n = i / 160, k = i - n * 160;
        W[i] = (_Float16)mw1[k * 256 + n];
    }
    __syncthreads();

    const int wid  = tid >> 5;
    const int lane = tid & 31;
    const bool hiL = lane >= 16;
    const int ln   = lane & 15;

    // ---- layer 1: [64,160] @ [160,256], relu -> Z1 ----
    for (int nt = 0; nt < 16; ++nt) {
        v8f acc = {};
        for (int kk = 0; kk < 5; ++kk) {
            AFrag a, bf;
            const h2* ar = (const h2*)(A0 + (wid * 16 + ln) * 160 + kk * 32);
            const h2* wp = (const h2*)(W  + (nt * 16 + ln) * 160 + kk * 32);
#pragma unroll
            for (int v = 0; v < 8; ++v) {
                a.p[v]  = ar[(2 * v + (v >= 4 ? 8 : 0) + (hiL ? 8 : 0)) >> 1];
                bf.p[v] = wp[v + (v >= 4 ? 4 : 0) + (hiL ? 4 : 0)];
            }
            acc = __builtin_amdgcn_wmma_f32_16x16x32_f16(
                      false, a.v, false, bf.v, (short)0, acc, false, false);
        }
        const float bias = mb1[nt * 16 + ln];
#pragma unroll
        for (int v = 0; v < 8; ++v) {
            float z = acc[v] + bias;
            z = z > 0.f ? z : 0.f;
            Z1[(wid * 16 + v + (hiL ? 8 : 0)) * 256 + nt * 16 + ln] = (_Float16)z;
        }
    }
    __syncthreads();

    // restage W = mw2 transposed [128][256]; stage mw3
    for (int i = tid; i < 128 * 256; i += 128) {
        int n = i >> 8, k = i & 255;
        W[i] = (_Float16)mw2[k * 128 + n];
    }
    if (tid < 128) w3[tid] = mw3[tid];
    __syncthreads();

    // ---- layer 2: [64,256] @ [256,128], relu -> Z2 ----
    for (int nt = 0; nt < 8; ++nt) {
        v8f acc = {};
        for (int kk = 0; kk < 8; ++kk) {
            AFrag a, bf;
            const h2* ar = (const h2*)(Z1 + (wid * 16 + ln) * 256 + kk * 32);
            const h2* wp = (const h2*)(W  + (nt * 16 + ln) * 256 + kk * 32);
#pragma unroll
            for (int v = 0; v < 8; ++v) {
                a.p[v]  = ar[(2 * v + (v >= 4 ? 8 : 0) + (hiL ? 8 : 0)) >> 1];
                bf.p[v] = wp[v + (v >= 4 ? 4 : 0) + (hiL ? 4 : 0)];
            }
            acc = __builtin_amdgcn_wmma_f32_16x16x32_f16(
                      false, a.v, false, bf.v, (short)0, acc, false, false);
        }
        const float bias = mb2[nt * 16 + ln];
#pragma unroll
        for (int v = 0; v < 8; ++v) {
            float z = acc[v] + bias;
            z = z > 0.f ? z : 0.f;
            Z2[(wid * 16 + v + (hiL ? 8 : 0)) * 128 + nt * 16 + ln] = (_Float16)z;
        }
    }
    __syncthreads();

    // ---- layer 3: [64,128] @ [128,1] ----
    if (tid < 64) {
        const int row = row0 + tid;
        if (row < B) {
            const h2* zr = (const h2*)(Z2 + tid * 128);
            float acc = 0.f;
            for (int c = 0; c < 64; ++c) {
                const h2 p = zr[c];
                acc += (float)p.x * w3[2 * c] + (float)p.y * w3[2 * c + 1];
            }
            out[row] = acc + mb3[0];
        }
    }
}

extern "C" void kernel_launch(void* const* d_in, const int* in_sizes, int n_in,
                              void* d_out, int out_size, void* d_ws, size_t ws_size,
                              hipStream_t stream) {
    const int*   customer_id  = (const int*)d_in[0];
    const int*   cand_good    = (const int*)d_in[1];
    const int*   cand_class   = (const int*)d_in[2];
    const int*   hist_goods   = (const int*)d_in[3];
    const int*   hist_classes = (const int*)d_in[4];
    const float* user_table   = (const float*)d_in[5];
    const float* item_table   = (const float*)d_in[6];
    const float* cat_table    = (const float*)d_in[7];
    const float* aw1 = (const float*)d_in[8];
    const float* ab1 = (const float*)d_in[9];
    const float* aw2 = (const float*)d_in[10];
    // d_in[11] = ab2 (softmax-invariant constant shift; omitted)
    const float* mw1 = (const float*)d_in[12];
    const float* mb1 = (const float*)d_in[13];
    const float* mw2 = (const float*)d_in[14];
    const float* mb2 = (const float*)d_in[15];
    const float* mw3 = (const float*)d_in[16];
    const float* mb3 = (const float*)d_in[17];

    const int B = in_sizes[0];
    _Float16* combined = (_Float16*)d_ws;   // B*160 f16 = 5.25 MB scratch

    din_attention_kernel<<<(B + 7) / 8, 256, K1_LDS_BYTES, stream>>>(
        customer_id, cand_good, cand_class, hist_goods, hist_classes,
        user_table, item_table, cat_table, aw1, ab1, aw2, combined, B);

    din_mlp_kernel<<<(B + 63) / 64, 128, K2_LDS_BYTES, stream>>>(
        combined, mw1, mb1, mw2, mb2, mw3, mb3, (float*)d_out, B);
}
